// SemanticNoveltyTracker_70205535420688
// MI455X (gfx1250) — matrix operand order repaired and checked by reading, not compiled, not run on hardware
//
#include <hip/hip_runtime.h>
#include <hip/hip_bf16.h>

// Problem constants (from reference)
#define BB   262144      // batch
#define KK   100         // clusters
#define KP   112         // clusters padded to 7*16
#define KT   7           // 16-wide k tiles
#define DD   384         // embedding dim
#define MOM  0.95f
#define EPSN 1e-8f

// LDS centroid row stride (bf16 elems), padded to avoid 16-way bank conflicts:
// 392*2 = 784 B = 196 dwords; 196 % 64 = 4 -> column-lanes spread across banks.
#define CSTR 392
#define CENT_LDS_BYTES (KP * CSTR * 2)        // 87808
#define SEG_LDS_BYTES  ((KK * DD + KK) * 4)   // 154000

typedef float  v8f    __attribute__((ext_vector_type(8)));
typedef __bf16 bf16x16 __attribute__((ext_vector_type(16)));

// ---------------------------------------------------------------------------
// K0b: normalize centroids (f32) -> unit bf16 rows in ws (padded rows = 0)
// ---------------------------------------------------------------------------
__global__ __launch_bounds__(128) void snt_centroid_prep(
    const float* __restrict__ cent, __bf16* __restrict__ centS)
{
    const int k = blockIdx.x;            // 0..111
    __shared__ float red[128];
    float vals[3];
    float s = 0.f;
    #pragma unroll
    for (int j = 0; j < 3; ++j) {
        const int d = threadIdx.x + j * 128;
        const float v = (k < KK) ? cent[k * DD + d] : 0.f;
        vals[j] = v;
        s = fmaf(v, v, s);
    }
    red[threadIdx.x] = s;
    __syncthreads();
    for (int off = 64; off > 0; off >>= 1) {
        if (threadIdx.x < off) red[threadIdx.x] += red[threadIdx.x + off];
        __syncthreads();
    }
    const float inv = 1.0f / fmaxf(sqrtf(red[0]), EPSN);
    #pragma unroll
    for (int j = 0; j < 3; ++j)
        centS[k * DD + threadIdx.x + j * 128] = (__bf16)(vals[j] * inv);
}

// ---------------------------------------------------------------------------
// K0a: zero segment accumulators in ws
// ---------------------------------------------------------------------------
__global__ void snt_zero(float* __restrict__ p, int n)
{
    const int i = blockIdx.x * 256 + threadIdx.x;
    if (i < n) p[i] = 0.f;
}

// ---------------------------------------------------------------------------
// K1: bf16 WMMA GEMM (e . c_hat) + per-row max/argmax -> novelty, closest
//   grid = B/128 WGs, 256 threads (8 waves), wave handles 16 rows x 112 cols
// ---------------------------------------------------------------------------
__global__ __launch_bounds__(256) void snt_gemm_argmax(
    const float* __restrict__ emb, const __bf16* __restrict__ centS,
    float* __restrict__ nov, float* __restrict__ clo_f)
{
    extern __shared__ char smem[];
    __bf16* cs = (__bf16*)smem;

    // Stage normalized bf16 centroids into LDS with padded row stride.
    {
        uint4* dst = (uint4*)smem;
        const uint4* src = (const uint4*)centS;
        // 112 rows * 48 uint4 per row (768 B of payload), dest stride 784 B (49 uint4)
        for (int i = threadIdx.x; i < KP * 48; i += 256) {
            const int r = i / 48, c = i % 48;
            dst[r * 49 + c] = src[r * 48 + c];
        }
    }
    __syncthreads();

    const int lane = threadIdx.x & 31;
    const int wave = threadIdx.x >> 5;
    const int half = lane >> 4;        // 0: K 0-7,16-23 ; 1: K 8-15,24-31 (A layout)
    const int col  = lane & 15;        // B/C column (N)
    const int base = blockIdx.x * 128 + wave * 16;
    const int myrow = base + col;
    const float* __restrict__ erow = emb + (size_t)myrow * DD;

    v8f acc[KT];
    {
        v8f z = {};
        #pragma unroll
        for (int kt = 0; kt < KT; ++kt) acc[kt] = z;
    }
    float nrm2 = 0.f;

    for (int d0 = 0; d0 < DD; d0 += 32) {
        __builtin_prefetch(erow + d0 + 32, 0, 1);  // global_prefetch next chunk
        // A fragment: row = myrow; this lane holds d0+half*8..+7 and d0+16+half*8..+7
        const float4* p0 = (const float4*)(erow + d0 + half * 8);
        const float4* p1 = (const float4*)(erow + d0 + 16 + half * 8);
        const float4 f0 = p0[0], f1 = p0[1], f2 = p1[0], f3 = p1[1];
        float t[16];
        t[0]=f0.x; t[1]=f0.y; t[2]=f0.z; t[3]=f0.w;
        t[4]=f1.x; t[5]=f1.y; t[6]=f1.z; t[7]=f1.w;
        t[8]=f2.x; t[9]=f2.y; t[10]=f2.z; t[11]=f2.w;
        t[12]=f3.x; t[13]=f3.y; t[14]=f3.z; t[15]=f3.w;
        bf16x16 a;
        #pragma unroll
        for (int i = 0; i < 16; ++i) {
            a[i] = (__bf16)t[i];
            nrm2 = fmaf(t[i], t[i], nrm2);
        }
        #pragma unroll
        for (int kt = 0; kt < KT; ++kt) {
            // B fragment: lane = column (cluster kt*16+col), d0+half*8..+7 and d0+16+half*8..+7
            union { uint4 u[2]; bf16x16 v; } b;
            const uint4* bp =
                (const uint4*)(cs + (kt * 16 + col) * CSTR + d0 + half * 8);
            b.u[0] = bp[0];
            b.u[1] = bp[2];
            acc[kt] = __builtin_amdgcn_wmma_f32_16x16x32_bf16(
                false, a, false, b.v, (short)0, acc[kt], false, false);
        }
    }

    // Full ||e_row||^2: lane l and l^16 hold complementary d-halves.
    nrm2 += __shfl_xor(nrm2, 16, 32);
    const float inv = 1.0f / fmaxf(sqrtf(nrm2), EPSN);

    // C layout: VGPR v -> row (v + half*8), N = col. Reduce argmax across 16 lanes.
    #pragma unroll
    for (int v = 0; v < 8; ++v) {
        float bd = -__builtin_inff();
        int   bk = 0;
        #pragma unroll
        for (int kt = 0; kt < KT; ++kt) {
            const int k = kt * 16 + col;
            const float dv = acc[kt][v];
            if (k < KK && dv > bd) { bd = dv; bk = k; }
        }
        #pragma unroll
        for (int m = 1; m < 16; m <<= 1) {
            const float od = __shfl_xor(bd, m, 32);
            const int   ok = __shfl_xor(bk, m, 32);
            if (od > bd || (od == bd && ok < bk)) { bd = od; bk = ok; }
        }
        // inv-norm of the row this VGPR belongs to lives on lane (v + half*8)
        const float invV = __shfl(inv, v + half * 8, 32);
        if (col == 0) {
            const int r = base + v + half * 8;
            const float cosv = bd * invV;
            nov[r]   = 1.0f - cosv * cosv;
            clo_f[r] = (float)bk;
        }
    }
}

// ---------------------------------------------------------------------------
// K2: segment sum via LDS [K][D] accumulation + global f32 atomic flush
//   grid = 128 WGs, each WG handles 2048 rows (wave = 256 rows)
// ---------------------------------------------------------------------------
__global__ __launch_bounds__(256) void snt_segsum(
    const float* __restrict__ emb, const float* __restrict__ clo_f,
    float* __restrict__ seg_g, float* __restrict__ cnt_g)
{
    extern __shared__ char smem[];
    float* seg = (float*)smem;       // K*D
    float* cnt = seg + KK * DD;      // K
    for (int i = threadIdx.x; i < KK * DD + KK; i += 256) seg[i] = 0.f;
    __syncthreads();

    const int lane = threadIdx.x & 31;
    const int wave = threadIdx.x >> 5;
    const int rbase = blockIdx.x * 2048 + wave * 256;

    for (int rr = 0; rr < 256; ++rr) {
        const int row = rbase + rr;
        const int c = (int)clo_f[row];
        const float* __restrict__ erow = emb + (size_t)row * DD;
        float* __restrict__ srow = seg + c * DD;
        #pragma unroll
        for (int i = 0; i < DD / 32; ++i) {
            const int d = i * 32 + lane;          // coalesced load, distinct LDS banks
            atomicAdd(&srow[d], erow[d]);         // ds_add_f32
        }
        if (lane == 0) atomicAdd(&cnt[c], 1.0f);
    }
    __syncthreads();

    for (int i = threadIdx.x; i < KK * DD; i += 256)
        unsafeAtomicAdd(&seg_g[i], seg[i]);       // global_atomic_add_f32
    for (int i = threadIdx.x; i < KK; i += 256)
        unsafeAtomicAdd(&cnt_g[i], cnt[i]);
}

// ---------------------------------------------------------------------------
// K3: EMA centroid update + counts
// ---------------------------------------------------------------------------
__global__ __launch_bounds__(256) void snt_finalize(
    const float* __restrict__ cent, const float* __restrict__ counts_in,
    const float* __restrict__ seg_g, const float* __restrict__ cnt_g,
    float* __restrict__ newc, float* __restrict__ newcnt)
{
    const int i = blockIdx.x * 256 + threadIdx.x;
    if (i < KK * DD) {
        const int k = i / DD;
        const float c = cnt_g[k];
        const float mean = seg_g[i] / fmaxf(c, 1.0f);
        const float oc = cent[i];
        newc[i] = (c > 0.f) ? MOM * oc + (1.0f - MOM) * mean : oc;
    }
    if (i < KK) newcnt[i] = counts_in[i] + cnt_g[i];
}

// ---------------------------------------------------------------------------
extern "C" void kernel_launch(void* const* d_in, const int* in_sizes, int n_in,
                              void* d_out, int out_size, void* d_ws, size_t ws_size,
                              hipStream_t stream)
{
    (void)in_sizes; (void)n_in; (void)out_size; (void)ws_size;

    const float* emb    = (const float*)d_in[0];   // [B, D]
    const float* cent   = (const float*)d_in[1];   // [K, D]
    const float* counts = (const float*)d_in[2];   // [K]

    float* out    = (float*)d_out;
    float* nov    = out;                    // [B]
    float* clo_f  = out + BB;               // [B] (argmax as float)
    float* newc   = out + 2 * BB;           // [K, D]
    float* newcnt = out + 2 * BB + KK * DD; // [K]

    // Workspace layout
    char* ws = (char*)d_ws;
    __bf16* centS = (__bf16*)ws;                         // KP*DD bf16 = 86016 B
    float*  seg_g = (float*)(ws + 86016);                // K*D f32   = 153600 B
    float*  cnt_g = (float*)(ws + 86016 + KK * DD * 4);  // K f32     = 400 B

    snt_centroid_prep<<<KP, 128, 0, stream>>>(cent, centS);
    snt_zero<<<(KK * DD + KK + 255) / 256, 256, 0, stream>>>(seg_g, KK * DD + KK);
    snt_gemm_argmax<<<BB / 128, 256, CENT_LDS_BYTES, stream>>>(emb, centS, nov, clo_f);
    snt_segsum<<<BB / 2048, 256, SEG_LDS_BYTES, stream>>>(emb, clo_f, seg_g, cnt_g);
    snt_finalize<<<(KK * DD + 255) / 256, 256, 0, stream>>>(cent, counts, seg_g, cnt_g,
                                                            newc, newcnt);
}
